// SelfAttention_35742717837814
// MI455X (gfx1250) — matrix operand order repaired and checked by reading, not compiled, not run on hardware
//
#include <hip/hip_runtime.h>
#include <math.h>

// ---------------------------------------------------------------------------
// Shapes (fixed by the reference)
// ---------------------------------------------------------------------------
constexpr int Bd = 2, Td = 2048, Cd = 1024, Hd = 16, Dd = 64;
constexpr int N3C  = 3 * Cd;        // 3072
constexpr int MTOK = Bd * Td;       // 4096
constexpr size_t HEADSZ = (size_t)Bd * Hd * Td * Dd; // elems per q/k/v tensor

typedef __attribute__((ext_vector_type(16))) __bf16 v16bf;
typedef __attribute__((ext_vector_type(8)))  float  v8f;
typedef __attribute__((ext_vector_type(4)))  unsigned int u32x4;
typedef __attribute__((ext_vector_type(8)))  int i32x8;
typedef __attribute__((ext_vector_type(4)))  int i32x4;

union V16 {
    v16bf v;
    unsigned short us[16];
    unsigned u[8];
};

__device__ __forceinline__ unsigned short f32_to_bf16(float f) {
    union { float f; unsigned u; } x; x.f = f;
    unsigned r = x.u + 0x7FFFu + ((x.u >> 16) & 1u);   // round-to-nearest-even
    return (unsigned short)(r >> 16);
}

// CDNA5 16-bit A-matrix 16x32 layout: lane 0-15 (half=0) holds K {0..7,16..23},
// lanes 16-31 (half=1) hold K {8..15,24..31}; 2 bf16 per VGPR.
__device__ __forceinline__ int a_kbase(int v, int half) {
    return ((v < 4) ? (2 * v) : (16 + 2 * (v - 4))) + 8 * half;
}
// CDNA5 16-bit B-matrix 32x16 layout: lane = column; same K order as A:
// lanes 0-15 hold K {0..7,16..23}, lanes 16-31 hold K {8..15,24..31}.
__device__ __forceinline__ int b_k0(int v, int half) {
    return ((v < 4) ? (2 * v) : (16 + 2 * (v - 4))) + 8 * half;
}

// ---------------------------------------------------------------------------
// TDM: async DMA of a 32(row) x 64(col) bf16 tile from a [2048 x 64] bf16
// tensor into LDS.  2D D# per CDNA5 ISA 8.3/8.4:
//   group0: count=1, lds_addr[63:32], global_addr[120:64], type=2
//   group1: data_size=2B, tensor_dim0=64, tensor_dim1=2048, tile=64x32,
//           tensor_dim0_stride=64
// Tracked by TENSORcnt; EXEC ignored (one DMA per wave).
// This toolchain exposes the 6-arg builtin:
//   (u32x4 g0, i32x8 g1, i32x4 g2, i32x4 g3, i32x8 g4, i32 cpol)
// ---------------------------------------------------------------------------
#if __has_builtin(__builtin_amdgcn_tensor_load_to_lds)
#define HAVE_TDM 1
__device__ __forceinline__ void tdm_load_tile32x64(const unsigned short* gbase,
                                                   unsigned lds_off) {
    unsigned long long ga = (unsigned long long)(uintptr_t)gbase;
    u32x4 g0;
    g0[0] = 1u;                                   // count=1, no gather
    g0[1] = lds_off;                              // LDS byte address
    g0[2] = (unsigned)ga;                         // global_addr[31:0]
    g0[3] = (unsigned)((ga >> 32) & 0x01FFFFFFu)  // global_addr[56:32]
            | (2u << 30);                         // type=2 ("image")
    i32x8 g1;
    g1[0] = 0x00010000;          // wg_mask=0, data_size=1 (2 bytes)
    g1[1] = (int)(64u << 16);    // tensor_dim0 = 64
    g1[2] = (int)(2048u << 16);  // tensor_dim1 = 2048
    g1[3] = (int)(64u << 16);    // tile_dim0 = 64
    g1[4] = 32;                  // tile_dim1 = 32, tile_dim2 = 0
    g1[5] = 64;                  // tensor_dim0_stride = 64
    g1[6] = 0;
    g1[7] = 0;
    i32x4 gz4 = {0, 0, 0, 0};
    i32x8 gz8 = {0, 0, 0, 0, 0, 0, 0, 0};
    __builtin_amdgcn_tensor_load_to_lds(g0, g1, gz4, gz4, gz8, 0);
}
#else
#define HAVE_TDM 0
#endif

// ---------------------------------------------------------------------------
// fp32 -> bf16 conversion (elementwise, memory bound, trivial)
// ---------------------------------------------------------------------------
__global__ void cvt_bf16_kernel(const float* __restrict__ in,
                                unsigned short* __restrict__ out, int n) {
    for (int i = blockIdx.x * blockDim.x + threadIdx.x; i < n;
         i += gridDim.x * blockDim.x)
        out[i] = f32_to_bf16(in[i]);
}

// ---------------------------------------------------------------------------
// WMMA GEMM: D = A[MxK] * B[KxN] + bias[N]
//   MODE 0: epilogue scatters bf16 result into q/k/v [B,H,T,D] head-major
//   MODE 1: epilogue writes fp32 result to out (proj)
// One wave computes one 16x16 tile; 8 waves/block -> 64m x 32n block tile.
// ---------------------------------------------------------------------------
template <int MODE>
__global__ void __launch_bounds__(256)
gemm_wmma_bf16(const unsigned short* __restrict__ A,
               const unsigned short* __restrict__ Bm,
               const float* __restrict__ bias,
               float* __restrict__ outf,
               unsigned short* __restrict__ outq,
               int M, int N, int K) {
    const int lane = threadIdx.x & 31;
    const int wave = threadIdx.x >> 5;
    const int half = lane >> 4;
    const int nl   = lane & 15;
    const int m0 = blockIdx.y * 64 + (wave >> 1) * 16;
    const int n0 = blockIdx.x * 32 + (wave & 1) * 16;

    const unsigned* Au = (const unsigned*)A;
    v8f acc = {};

    for (int kk = 0; kk < K; kk += 32) {
        V16 a, bt;
#pragma unroll
        for (int v = 0; v < 8; ++v) {
            const int ka = kk + a_kbase(v, half);
            a.u[v] = Au[(((size_t)(m0 + nl)) * K + ka) >> 1];
            const int kb = kk + b_k0(v, half);
            bt.us[2 * v]     = Bm[(size_t)kb * N + n0 + nl];
            bt.us[2 * v + 1] = Bm[(size_t)(kb + 1) * N + n0 + nl];
        }
        if (kk + 64 < K)  // gfx1250 global_prefetch_b8 hint for the A tile two k-steps ahead
            __builtin_prefetch((const void*)&Au[(((size_t)(m0 + nl)) * K + kk + 64) >> 1], 0, 0);
        acc = __builtin_amdgcn_wmma_f32_16x16x32_bf16(
            false, a.v, false, bt.v, (short)0, acc, false, false);
    }

    const int j  = n0 + nl;
    const float bj = bias[j];
#pragma unroll
    for (int r = 0; r < 8; ++r) {
        const int g = m0 + r + 8 * half;        // global token row
        const float val = acc[r] + bj;
        if (MODE == 0) {
            const int sel = j / Cd, rem = j % Cd;
            const int hh = rem / Dd, dd = rem % Dd;
            const int bb = g >> 11, t = g & (Td - 1);
            const size_t off = (size_t)sel * HEADSZ +
                               (((size_t)(bb * Hd + hh)) * Td + t) * Dd + dd;
            outq[off] = f32_to_bf16(val);
        } else {
            outf[(size_t)g * N + j] = val;
        }
    }
}

// ---------------------------------------------------------------------------
// Causal flash attention. Grid: (T/64, B*H), 128 threads (4 waves).
// Each wave owns 16 queries; loops 32-key blocks with online softmax.
// K/V tiles are DMA'd into per-wave double-buffered LDS by the Tensor Data
// Mover (TENSORcnt-pipelined: next tile's DMA overlaps current tile's WMMAs).
// S = Q K^T via 2x wmma (K-dim 64 = 2x32), P staged through LDS
// (C-layout -> A-layout transpose), O += P V via 4x wmma (Dd = 4x16 n-tiles).
// ---------------------------------------------------------------------------
__global__ void __launch_bounds__(128)
attn_kernel(const unsigned short* __restrict__ qkv,
            unsigned short* __restrict__ y) {
    __shared__ unsigned short ldsK[4][2][32 * 64];  // 32 KiB
    __shared__ unsigned short ldsV[4][2][32 * 64];  // 32 KiB
    __shared__ unsigned short ldsP[4][16 * 32];     //  4 KiB

    const int lane = threadIdx.x & 31;
    const int wave = threadIdx.x >> 5;
    const int half = lane >> 4;
    const int nl   = lane & 15;
    const int bh = blockIdx.y;
    const int b = bh / Hd, h = bh % Hd;
    const int q0 = blockIdx.x * 64 + wave * 16;

    const size_t headoff = ((size_t)(b * Hd + h)) * Td * Dd;
    const unsigned short* Kg = qkv + HEADSZ + headoff;
    const unsigned short* Vg = qkv + 2 * HEADSZ + headoff;
    const unsigned* Qu = (const unsigned*)(qkv + headoff);

    // Preload the Q A-tiles (16x32 each, d 0..31 and d 32..63)
    V16 aq0, aq1;
#pragma unroll
    for (int v = 0; v < 8; ++v) {
        const int ka = a_kbase(v, half);
        aq0.u[v] = Qu[((q0 + nl) * Dd + ka) >> 1];
        aq1.u[v] = Qu[((q0 + nl) * Dd + 32 + ka) >> 1];
    }

    float mrow[8], lrow[8];
    v8f Ot[4];
    const v8f zf = {};
#pragma unroll
    for (int r = 0; r < 8; ++r) { mrow[r] = -3.0e38f; lrow[r] = 0.0f; }
#pragma unroll
    for (int t = 0; t < 4; ++t) Ot[t] = zf;

    const float scale = 0.125f;  // 1/sqrt(64)
    const int kend = q0 + 16;    // causal bound (exclusive)

#if HAVE_TDM
    // Prologue: DMA first K/V tiles into buffer 0.
    tdm_load_tile32x64(Kg, (unsigned)(uintptr_t)&ldsK[wave][0][0]);
    tdm_load_tile32x64(Vg, (unsigned)(uintptr_t)&ldsV[wave][0][0]);
#endif

    for (int kb = 0; kb < kend; kb += 32) {
        const int buf = (kb >> 5) & 1;
#if HAVE_TDM
        // Pipeline: issue next tile's DMA before consuming the current one.
        if (kb + 32 < kend) {
            tdm_load_tile32x64(Kg + (size_t)(kb + 32) * Dd,
                               (unsigned)(uintptr_t)&ldsK[wave][buf ^ 1][0]);
            tdm_load_tile32x64(Vg + (size_t)(kb + 32) * Dd,
                               (unsigned)(uintptr_t)&ldsV[wave][buf ^ 1][0]);
            __builtin_amdgcn_s_wait_tensorcnt(2);  // current 2 DMAs complete
        } else {
            __builtin_amdgcn_s_wait_tensorcnt(0);
        }
        const unsigned* lk = (const unsigned*)&ldsK[wave][buf][0];
        const unsigned short* lv = &ldsV[wave][buf][0];
#else
        // Fallback: per-lane gathers straight from global.
        const unsigned* lk = (const unsigned*)(Kg + (size_t)kb * Dd);
        const unsigned short* lv = Vg + (size_t)kb * Dd;
#endif

        // --- B tiles from the K tile (B[d][key] = K[key][d]) ---
        V16 bk00, bk01, bk10, bk11;
#pragma unroll
        for (int v = 0; v < 8; ++v) {
            const int k0 = b_k0(v, half);
            bk00.u[v] = lk[(nl * Dd + k0) >> 1];
            bk01.u[v] = lk[(nl * Dd + 32 + k0) >> 1];
            bk10.u[v] = lk[((16 + nl) * Dd + k0) >> 1];
            bk11.u[v] = lk[((16 + nl) * Dd + 32 + k0) >> 1];
        }
        v8f s0 = __builtin_amdgcn_wmma_f32_16x16x32_bf16(
            false, aq0.v, false, bk00.v, (short)0, zf, false, false);
        s0 = __builtin_amdgcn_wmma_f32_16x16x32_bf16(
            false, aq1.v, false, bk01.v, (short)0, s0, false, false);
        v8f s1 = __builtin_amdgcn_wmma_f32_16x16x32_bf16(
            false, aq0.v, false, bk10.v, (short)0, zf, false, false);
        s1 = __builtin_amdgcn_wmma_f32_16x16x32_bf16(
            false, aq1.v, false, bk11.v, (short)0, s1, false, false);

        // --- causal mask + online softmax (row = r + 8*half, col = nl) ---
        float p0[8], p1[8];
#pragma unroll
        for (int r = 0; r < 8; ++r) {
            const int q = q0 + r + 8 * half;
            float v0 = s0[r] * scale, v1 = s1[r] * scale;
            if (kb + nl > q)      v0 = -3.0e38f;
            if (kb + 16 + nl > q) v1 = -3.0e38f;
            float mx = fmaxf(v0, v1);
#pragma unroll
            for (int off = 1; off < 16; off <<= 1)
                mx = fmaxf(mx, __shfl_xor(mx, off, 32));
            const float mn = fmaxf(mrow[r], mx);
            const float al = __expf(mrow[r] - mn);
            const float e0 = __expf(v0 - mn), e1 = __expf(v1 - mn);
            float sum = e0 + e1;
#pragma unroll
            for (int off = 1; off < 16; off <<= 1)
                sum += __shfl_xor(sum, off, 32);
            lrow[r] = lrow[r] * al + sum;
            mrow[r] = mn;
            p0[r] = e0; p1[r] = e1;
#pragma unroll
            for (int t = 0; t < 4; ++t) Ot[t][r] *= al;
        }

        // --- stage P (C-layout) into LDS, reload as 16x32 bf16 A-tile ---
#pragma unroll
        for (int r = 0; r < 8; ++r) {
            const int m = r + 8 * half;
            ldsP[wave][m * 32 + nl]      = f32_to_bf16(p0[r]);
            ldsP[wave][m * 32 + 16 + nl] = f32_to_bf16(p1[r]);
        }
        __asm__ volatile("" ::: "memory");  // same-wave DS ops are in-order
        V16 pa;
        const unsigned* lu = (const unsigned*)&ldsP[wave][0];
#pragma unroll
        for (int v = 0; v < 8; ++v)
            pa.u[v] = lu[(nl * 32 + a_kbase(v, half)) >> 1];

        // --- O[t] += P * V, per 16-wide d chunk (V read from LDS tile) ---
#pragma unroll
        for (int t = 0; t < 4; ++t) {
            V16 bv;
#pragma unroll
            for (int v = 0; v < 8; ++v) {
                const int k0 = b_k0(v, half);
                bv.us[2 * v]     = lv[(k0) * Dd + t * 16 + nl];
                bv.us[2 * v + 1] = lv[(k0 + 1) * Dd + t * 16 + nl];
            }
            Ot[t] = __builtin_amdgcn_wmma_f32_16x16x32_bf16(
                false, pa.v, false, bv.v, (short)0, Ot[t], false, false);
        }
    }

    // --- normalize and store y[B,T,C] as bf16 for the proj GEMM ---
#pragma unroll
    for (int t = 0; t < 4; ++t) {
#pragma unroll
        for (int r = 0; r < 8; ++r) {
            const int m = r + 8 * half;
            const float val = Ot[t][r] / lrow[r];
            const size_t row = (size_t)b * Td + q0 + m;
            y[row * Cd + h * Dd + t * 16 + nl] = f32_to_bf16(val);
        }
    }
}

// ---------------------------------------------------------------------------
// Orchestration
// ---------------------------------------------------------------------------
extern "C" void kernel_launch(void* const* d_in, const int* in_sizes, int n_in,
                              void* d_out, int out_size, void* d_ws, size_t ws_size,
                              hipStream_t stream) {
    const float* x      = (const float*)d_in[0];
    const float* w_qkv  = (const float*)d_in[1];
    const float* b_qkv  = (const float*)d_in[2];
    const float* w_proj = (const float*)d_in[3];
    const float* b_proj = (const float*)d_in[4];
    float* out = (float*)d_out;

    char* ws = (char*)d_ws;
    unsigned short* xb     = (unsigned short*)(ws);                 //  8 MiB
    unsigned short* wqkvb  = (unsigned short*)(ws + 8388608);       //  6 MiB
    unsigned short* wprojb = (unsigned short*)(ws + 14680064);      //  2 MiB
    unsigned short* qkvb   = (unsigned short*)(ws + 16777216);      // 24 MiB
    unsigned short* yb     = (unsigned short*)(ws + 41943040);      //  8 MiB

    // fp32 -> bf16 staging
    cvt_bf16_kernel<<<2048, 256, 0, stream>>>(x,      xb,     MTOK * Cd);
    cvt_bf16_kernel<<<2048, 256, 0, stream>>>(w_qkv,  wqkvb,  Cd * N3C);
    cvt_bf16_kernel<<<1024, 256, 0, stream>>>(w_proj, wprojb, Cd * Cd);

    // qkv = x @ w_qkv + b_qkv, scattered to [3][B,H,T,D] bf16
    dim3 gq(N3C / 32, MTOK / 64);
    gemm_wmma_bf16<0><<<gq, 256, 0, stream>>>(xb, wqkvb, b_qkv, nullptr, qkvb,
                                              MTOK, N3C, Cd);

    // causal flash attention (TDM-fed) -> yb [B,T,C] bf16
    dim3 ga(Td / 64, Bd * Hd);
    attn_kernel<<<ga, 128, 0, stream>>>(qkvb, yb);

    // out = yb @ w_proj + b_proj (fp32)
    dim3 gp(Cd / 32, MTOK / 64);
    gemm_wmma_bf16<1><<<gp, 256, 0, stream>>>(yb, wprojb, b_proj, out, nullptr,
                                              MTOK, Cd, Cd);
}
